// superpoint_transformer_55336358643430
// MI455X (gfx1250) — matrix operand order repaired and checked by reading, not compiled, not run on hardware
//
#include <hip/hip_runtime.h>
#include <math.h>

// ---------------------------------------------------------------------------
// Superpoint-transformer association kernel for gfx1250 (MI455X).
//   N=80000 points, K=8 neighbors, M=1000 superpoints, C=128, OUT=64, S=2.
// Strategy: precompute per-superpoint tables (kpe = K-proj + pos-enc,
// u = S-folded (V-proj + pos-enc)), then one fused WMMA kernel:
// block = 16 points (256 thr = 8 waves), wave = 2 points = 16 (n,k) rows.
// GEMMs run as v_wmma_f32_16x16x32_bf16 (f32 accumulate).
// ---------------------------------------------------------------------------

typedef __attribute__((ext_vector_type(16))) __bf16 v16bf;
typedef __attribute__((ext_vector_type(8)))  __bf16 v8bf;
typedef __attribute__((ext_vector_type(8)))  float  v8f;
typedef __attribute__((ext_vector_type(4)))  float  v4f;

#define EPS 1e-5f

// ---- Kernel 0: fold BatchNorm affines into per-channel scale/shift --------
__global__ void spt_setup(const float* __restrict__ g1, const float* __restrict__ beta1,
                          const float* __restrict__ m1, const float* __restrict__ v1,
                          const float* __restrict__ b1, const float* __restrict__ g2,
                          const float* __restrict__ beta2, const float* __restrict__ m2,
                          const float* __restrict__ v2,
                          float* __restrict__ s1, float* __restrict__ t1,
                          float* __restrict__ s2, float* __restrict__ t2p) {
  int c = threadIdx.x;
  if (c < 128) {
    float s = g1[c] * rsqrtf(v1[c] + EPS);
    s1[c] = s;
    t1[c] = beta1[c] - m1[c] * s;           // bn1(w) = s1*w + t1
  }
  if (c < 64) {
    float s = g2[c] * rsqrtf(v2[c] + EPS);
    s2[c] = s;
    t2p[c] = s * b1[c] + beta2[c] - m2[c] * s;  // bn2(h+b1) = s2*h + t2p
  }
}

// ---- Kernel 1: per-superpoint tables (M=1000 rows, trivial FLOPs) ---------
//   kpe[m][c] = (sp_fea@Wk + bk)[m][c] + pe[m][c]          (c in [0,128))
//   u[m][j]   = (vsp+pe)[m][j] + (vsp+pe)[m][j+64]         (j in [0,64))
__global__ void spt_sp_pre(const float* __restrict__ sp_fea, const float* __restrict__ sp_xyz,
                           const float* __restrict__ Wk, const float* __restrict__ bk,
                           const float* __restrict__ Wv, const float* __restrict__ bv,
                           const float* __restrict__ Wp1, const float* __restrict__ bp1,
                           const float* __restrict__ gp, const float* __restrict__ betap,
                           const float* __restrict__ mp, const float* __restrict__ vp,
                           const float* __restrict__ Wp2, const float* __restrict__ bp2,
                           float* __restrict__ kpe, float* __restrict__ u) {
  __shared__ float srow[128];
  __shared__ float svpe[128];
  const int row = blockIdx.x;
  const int c   = threadIdx.x;
  srow[c] = sp_fea[row * 128 + c];
  __syncthreads();

  // positional MLP: Linear(3,3) -> BN(3) -> ReLU -> Linear(3,128)
  float x0 = sp_xyz[row * 3 + 0], x1 = sp_xyz[row * 3 + 1], x2 = sp_xyz[row * 3 + 2];
  float h[3];
#pragma unroll
  for (int i = 0; i < 3; ++i) {
    float hv = x0 * Wp1[0 * 3 + i] + x1 * Wp1[1 * 3 + i] + x2 * Wp1[2 * 3 + i] + bp1[i];
    hv = (hv - mp[i]) * (gp[i] * rsqrtf(vp[i] + EPS)) + betap[i];
    h[i] = fmaxf(hv, 0.f);
  }
  float pe = h[0] * Wp2[0 * 128 + c] + h[1] * Wp2[1 * 128 + c] + h[2] * Wp2[2 * 128 + c] + bp2[c];

  float ks = bk[c], vs = bv[c];
  for (int j = 0; j < 128; ++j) {
    float f = srow[j];
    ks += f * Wk[j * 128 + c];
    vs += f * Wv[j * 128 + c];
  }
  kpe[row * 128 + c] = ks + pe;
  svpe[c] = vs + pe;
  __syncthreads();
  if (c < 64) u[row * 64 + c] = svpe[c] + svpe[c + 64];
}

// ---- Kernel 2: fused attention (block = 16 points, wave = 2 points) -------
__global__ __launch_bounds__(256) void spt_main(
    const float* __restrict__ o_p_fea, const float* __restrict__ Wq, const float* __restrict__ bq,
    const float* __restrict__ W1, const float* __restrict__ W2,
    const float* __restrict__ kpe, const float* __restrict__ u,
    const float* __restrict__ s1, const float* __restrict__ t1,
    const float* __restrict__ s2, const float* __restrict__ t2p,
    const int* __restrict__ idx_abs, float* __restrict__ out, int N) {
  __shared__ __bf16 sW1[128 * 64];   // 16 KB
  __shared__ __bf16 sW2[64 * 64];    //  8 KB
  __shared__ float  sQ[16 * 128];    //  8 KB (this block's 16 q rows)
  __shared__ __bf16 sH1[8 * 16 * 64];//  16 KB (per-wave H1 transpose staging)

  const int tid  = threadIdx.x;
  const int lane = tid & 31;
  const int wave = tid >> 5;
  const int lmod = lane & 15;   // col (B/C/D) or row (A) selector
  const int g    = lane >> 4;   // lane-half selector
  const int pt_base = blockIdx.x * 16;

  // stage W1/W2 into LDS as bf16 (cooperative, 256 threads)
  for (int i = tid; i < 128 * 64; i += 256) sW1[i] = (__bf16)W1[i];
  for (int i = tid; i < 64 * 64; i += 256)  sW2[i] = (__bf16)W2[i];

  const v8f zero8 = {0.f, 0.f, 0.f, 0.f, 0.f, 0.f, 0.f, 0.f};

  // ---- Phase Q: wave w computes q columns [16w,16w+16) for all 16 points
  {
    int prow = pt_base + lmod; if (prow >= N) prow = N - 1;
    const float* arow = o_p_fea + (long)prow * 128;
    v16bf afrag[4];
#pragma unroll
    for (int f = 0; f < 4; ++f) {
#pragma unroll
      for (int ch = 0; ch < 2; ++ch) {
        int c0 = 32 * f + 16 * ch + 8 * g;
        v4f a0 = *(const v4f*)(arow + c0);
        v4f a1 = *(const v4f*)(arow + c0 + 4);
#pragma unroll
        for (int j = 0; j < 4; ++j) {
          afrag[f][8 * ch + j]     = (__bf16)a0[j];
          afrag[f][8 * ch + 4 + j] = (__bf16)a1[j];
        }
      }
    }
    v8f acc = zero8;
#pragma unroll
    for (int f = 0; f < 4; ++f) {
      const float* wcol = Wq + (long)(32 * f) * 128 + 16 * wave + lmod;
      v16bf bfrag;
#pragma unroll
      for (int s = 0; s < 16; ++s) {
        int k = (s < 8) ? (8 * g + s) : (16 + 8 * g + (s - 8));
        bfrag[s] = (__bf16)wcol[(long)k * 128];
      }
      acc = __builtin_amdgcn_wmma_f32_16x16x32_bf16(false, afrag[f], false, bfrag,
                                                    (short)0, acc, false, false);
    }
    float bias = bq[16 * wave + lmod];
#pragma unroll
    for (int r = 0; r < 8; ++r)
      sQ[(r + 8 * g) * 128 + 16 * wave + lmod] = acc[r] + bias;
  }
  __syncthreads();

  // ---- Phase attention: this wave owns points [pt_base+2w, +2), 16 rows
  const int p_in_blk = wave * 2;
  int n_m = pt_base + p_in_blk + (lmod >> 3);         // point of row lmod
  int n_mc = n_m < N ? n_m : N - 1;
  int kk = lmod & 7;
  int sp = idx_abs[(long)n_mc * 8 + kk];              // superpoint for this row
  const float* krow = kpe + (long)sp * 128;
  const float* qrow = sQ + (p_in_blk + (lmod >> 3)) * 128;

  // Build A fragments of X = relu(s1*(kpe[idx]-q)+t1), bf16
  v16bf xfrag[4];
#pragma unroll
  for (int f = 0; f < 4; ++f) {
#pragma unroll
    for (int ch = 0; ch < 2; ++ch) {
      int c0 = 32 * f + 16 * ch + 8 * g;
      v4f k0 = *(const v4f*)(krow + c0), k1 = *(const v4f*)(krow + c0 + 4);
      v4f q0 = *(const v4f*)(qrow + c0), q1 = *(const v4f*)(qrow + c0 + 4);
      v4f a0 = *(const v4f*)(s1 + c0),   a1 = *(const v4f*)(s1 + c0 + 4);
      v4f b0 = *(const v4f*)(t1 + c0),   b1 = *(const v4f*)(t1 + c0 + 4);
#pragma unroll
      for (int j = 0; j < 4; ++j) {
        xfrag[f][8 * ch + j]     = (__bf16)fmaxf(a0[j] * (k0[j] - q0[j]) + b0[j], 0.f);
        xfrag[f][8 * ch + 4 + j] = (__bf16)fmaxf(a1[j] * (k1[j] - q1[j]) + b1[j], 0.f);
      }
    }
  }

  // H1 = X @ W1   (16x128 @ 128x64)
  v8f acc1[4];
#pragma unroll
  for (int nt = 0; nt < 4; ++nt) {
    acc1[nt] = zero8;
#pragma unroll
    for (int f = 0; f < 4; ++f) {
      v16bf bfrag;
#pragma unroll
      for (int s = 0; s < 16; ++s)
        bfrag[s] = sW1[(32 * f + ((s < 8) ? (8 * g + s) : (16 + 8 * g + (s - 8)))) * 64 + 16 * nt + lmod];
      acc1[nt] = __builtin_amdgcn_wmma_f32_16x16x32_bf16(false, xfrag[f], false, bfrag,
                                                         (short)0, acc1[nt], false, false);
    }
  }

  // bn2 + relu, stage bf16 to per-wave LDS tile (accumulator->A transpose)
  __bf16* myH1 = sH1 + wave * 16 * 64;
#pragma unroll
  for (int nt = 0; nt < 4; ++nt) {
    int col = 16 * nt + lmod;
    float sc = s2[col], sh = t2p[col];
#pragma unroll
    for (int r = 0; r < 8; ++r)
      myH1[(r + 8 * g) * 64 + col] = (__bf16)fmaxf(sc * acc1[nt][r] + sh, 0.f);
  }
  // wave-private staging: LDS is in-order per wave; drain DScnt before reads
  asm volatile("s_wait_dscnt 0" ::: "memory");

  // A2 fragments from staged H1 (16x64)
  v16bf a2[2];
#pragma unroll
  for (int f2 = 0; f2 < 2; ++f2) {
#pragma unroll
    for (int ch = 0; ch < 2; ++ch) {
      int c0 = 32 * f2 + 16 * ch + 8 * g;
      v8bf chunk = *(const v8bf*)&myH1[lmod * 64 + c0];
#pragma unroll
      for (int j = 0; j < 8; ++j) a2[f2][8 * ch + j] = chunk[j];
    }
  }

  // H2 = A2 @ W2   (16x64 @ 64x64). b2 omitted: per-column constant cancels
  // in the softmax over K (axis=1).
  v8f acc2[4];
#pragma unroll
  for (int nt = 0; nt < 4; ++nt) {
    acc2[nt] = zero8;
#pragma unroll
    for (int f2 = 0; f2 < 2; ++f2) {
      v16bf bfrag;
#pragma unroll
      for (int s = 0; s < 16; ++s)
        bfrag[s] = sW2[(32 * f2 + ((s < 8) ? (8 * g + s) : (16 + 8 * g + (s - 8)))) * 64 + 16 * nt + lmod];
      acc2[nt] = __builtin_amdgcn_wmma_f32_16x16x32_bf16(false, a2[f2], false, bfrag,
                                                         (short)0, acc2[nt], false, false);
    }
  }

  // Each lane now holds, for point (pt_base + 2w + g) and column 16nt+lmod,
  // all K=8 neighbor values in acc2[nt][0..7] -> lane-local softmax over K.
  int n_pt = pt_base + p_in_blk + g;
  int n_pt_c = n_pt < N ? n_pt : N - 1;
  int spi[8];
#pragma unroll
  for (int r = 0; r < 8; ++r) spi[r] = idx_abs[(long)n_pt_c * 8 + r];

  float bi[8];
#pragma unroll
  for (int r = 0; r < 8; ++r) bi[r] = 0.f;

#pragma unroll
  for (int nt = 0; nt < 4; ++nt) {
    int col = 16 * nt + lmod;
    float mx = acc2[nt][0];
#pragma unroll
    for (int r = 1; r < 8; ++r) mx = fmaxf(mx, acc2[nt][r]);
    float e[8], ssum = 0.f;
#pragma unroll
    for (int r = 0; r < 8; ++r) { e[r] = __expf(acc2[nt][r] - mx); ssum += e[r]; }
    float inv = 1.f / ssum;
#pragma unroll
    for (int r = 0; r < 8; ++r)
      bi[r] += (e[r] * inv) * u[(long)spi[r] * 64 + col];
  }

  // reduce the 64 columns: xor-shuffle across the 16 lanes of this g-half
#pragma unroll
  for (int msk = 1; msk < 16; msk <<= 1) {
#pragma unroll
    for (int r = 0; r < 8; ++r) bi[r] += __shfl_xor(bi[r], msk, 32);
  }

  // final softmax over K; one writer per g-half
  if (lmod == 0 && n_pt < N) {
    float mx = bi[0];
#pragma unroll
    for (int r = 1; r < 8; ++r) mx = fmaxf(mx, bi[r]);
    float e[8], ssum = 0.f;
#pragma unroll
    for (int r = 0; r < 8; ++r) { e[r] = __expf(bi[r] - mx); ssum += e[r]; }
    float inv = 1.f / ssum;
#pragma unroll
    for (int r = 0; r < 8; ++r) out[(long)n_pt * 8 + r] = e[r] * inv;
  }
}

// ---------------------------------------------------------------------------
extern "C" void kernel_launch(void* const* d_in, const int* in_sizes, int n_in,
                              void* d_out, int out_size, void* d_ws, size_t ws_size,
                              hipStream_t stream) {
  const float* sp_fea  = (const float*)d_in[0];
  const float* sp_xyz  = (const float*)d_in[1];
  const float* o_p_fea = (const float*)d_in[2];
  // d_in[3] = p_xyz (unused by the reference)
  const float* Wq  = (const float*)d_in[4];  const float* bq  = (const float*)d_in[5];
  const float* Wk  = (const float*)d_in[6];  const float* bk  = (const float*)d_in[7];
  const float* Wv  = (const float*)d_in[8];  const float* bv  = (const float*)d_in[9];
  const float* Wp1 = (const float*)d_in[10]; const float* bp1 = (const float*)d_in[11];
  const float* gp  = (const float*)d_in[12]; const float* betap = (const float*)d_in[13];
  const float* mp  = (const float*)d_in[14]; const float* vp  = (const float*)d_in[15];
  const float* Wp2 = (const float*)d_in[16]; const float* bp2 = (const float*)d_in[17];
  const float* g1  = (const float*)d_in[18]; const float* beta1 = (const float*)d_in[19];
  const float* m1  = (const float*)d_in[20]; const float* v1  = (const float*)d_in[21];
  const float* W1  = (const float*)d_in[22]; const float* b1  = (const float*)d_in[23];
  const float* g2  = (const float*)d_in[24]; const float* beta2 = (const float*)d_in[25];
  const float* m2  = (const float*)d_in[26]; const float* v2  = (const float*)d_in[27];
  const float* W2  = (const float*)d_in[28];
  // d_in[29] = b2 (provably cancels in softmax over K)
  const int* idx_abs = (const int*)d_in[30];
  // d_in[31..34] unused by the reference

  const int M = in_sizes[0] / 128;   // superpoints
  const int N = in_sizes[2] / 128;   // points

  // workspace layout (floats): needs (M*192 + 384)*4 bytes (~0.77 MB for M=1000)
  float* ws  = (float*)d_ws;
  float* kpe = ws;                       // [M,128]
  float* u   = kpe + (size_t)M * 128;    // [M,64]
  float* s1  = u + (size_t)M * 64;       // [128]
  float* t1  = s1 + 128;                 // [128]
  float* s2  = t1 + 128;                 // [64]
  float* t2p = s2 + 64;                  // [64]

  spt_setup<<<1, 128, 0, stream>>>(g1, beta1, m1, v1, b1, g2, beta2, m2, v2,
                                   s1, t1, s2, t2p);
  spt_sp_pre<<<M, 128, 0, stream>>>(sp_fea, sp_xyz, Wk, bk, Wv, bv, Wp1, bp1,
                                    gp, betap, mp, vp, Wp2, bp2, kpe, u);
  spt_main<<<(N + 15) / 16, 256, 0, stream>>>(o_p_fea, Wq, bq, W1, W2, kpe, u,
                                              s1, t1, s2, t2p, idx_abs,
                                              (float*)d_out, N);
}
